// GQAAttention_10711648436332
// MI455X (gfx1250) — matrix operand order repaired and checked by reading, not compiled, not run on hardware
//
#include <hip/hip_runtime.h>

typedef __attribute__((ext_vector_type(16))) __bf16          v16bf;
typedef __attribute__((ext_vector_type(8)))  float           v8f;
typedef __attribute__((ext_vector_type(8)))  unsigned short  us8;
typedef __attribute__((ext_vector_type(16))) unsigned short  us16;

#define BB     2
#define T_SEQ  2048
#define CDIM   2048
#define NKV    2
#define RQ     8
#define DH     128

// ---------- helpers ----------
__device__ __forceinline__ unsigned short f2bf(float f) {
    unsigned int u = __builtin_bit_cast(unsigned int, f);
    u += 0x7fffu + ((u >> 16) & 1u);        // round-to-nearest-even
    return (unsigned short)(u >> 16);
}

__device__ __forceinline__ v16bf mkfrag(us8 lo, us8 hi) {
    union { us16 v; us8 h[2]; } u;
    u.h[0] = lo; u.h[1] = hi;
    return __builtin_bit_cast(v16bf, u.v);
}

__device__ __forceinline__ v16bf ldfrag(const unsigned short* p0, const unsigned short* p1) {
    return mkfrag(*(const us8*)p0, *(const us8*)p1);
}

__device__ __forceinline__ v8f wmma_bf16(v16bf a, v16bf b, v8f c) {
    return __builtin_amdgcn_wmma_f32_16x16x32_bf16(false, a, false, b, (short)0, c, false, false);
}

// ---------- prep kernels ----------
__global__ void cvt_bf16(const float* __restrict__ in, unsigned short* __restrict__ out, int n) {
    for (int i = blockIdx.x * blockDim.x + threadIdx.x; i < n; i += gridDim.x * blockDim.x)
        out[i] = f2bf(in[i]);
}

// in: [K,N] fp32 row-major -> out: [N,K] bf16 row-major ; 32x32 LDS tiles, 256 threads
__global__ void transpose_cvt(const float* __restrict__ in, unsigned short* __restrict__ out,
                              int K, int N) {
    __shared__ unsigned short tile[32][33];
    const int k0 = blockIdx.x * 32, n0 = blockIdx.y * 32;
    const int tx = threadIdx.x & 31, ty = threadIdx.x >> 5;   // ty in 0..7
#pragma unroll
    for (int i = 0; i < 4; ++i)                                // coalesced read along N
        tile[ty + i * 8][tx] = f2bf(in[(size_t)(k0 + ty + i * 8) * N + n0 + tx]);
    __syncthreads();
#pragma unroll
    for (int i = 0; i < 4; ++i)                                // coalesced write along K
        out[(size_t)(n0 + ty + i * 8) * K + k0 + tx] = tile[tx][ty + i * 8];
}

// Vb: [B*T, NKV*DH] bf16 -> Vt: [B][NKV*DH][T] bf16 (per-batch transpose, 32x32 tiles)
__global__ void vtrans(const unsigned short* __restrict__ Vb, unsigned short* __restrict__ Vt) {
    __shared__ unsigned short tile[32][33];
    const int b = blockIdx.z;
    const int t0 = blockIdx.x * 32, d0 = blockIdx.y * 32;
    const int tx = threadIdx.x & 31, ty = threadIdx.x >> 5;
#pragma unroll
    for (int i = 0; i < 4; ++i)
        tile[ty + i * 8][tx] = Vb[(size_t)(b * T_SEQ + t0 + ty + i * 8) * (NKV * DH) + d0 + tx];
    __syncthreads();
#pragma unroll
    for (int i = 0; i < 4; ++i)
        Vt[((size_t)b * NKV * DH + d0 + ty + i * 8) * T_SEQ + t0 + tx] = tile[tx][ty + i * 8];
}

// ---------- generic bf16 WMMA GEMM: C[M,N] = A[M,K] @ Bt[N,K]^T ----------
template <bool OUT_BF16>
__global__ void gemm_wmma(const unsigned short* __restrict__ A,
                          const unsigned short* __restrict__ Bt,
                          float* __restrict__ Cf,
                          unsigned short* __restrict__ Cb,
                          int M, int N, int Kd) {
    const int lane = threadIdx.x & 31;
    const int ln = lane & 15, lg = lane >> 4;
    const int wid = blockIdx.x * (blockDim.x >> 5) + (threadIdx.x >> 5);
    const int tilesN = N >> 6;
    const int m0 = (wid / tilesN) << 4;
    const int n0 = (wid % tilesN) << 6;
    if (m0 >= M) return;

    v8f acc[4] = {{}, {}, {}, {}};
    const unsigned short* arow = A + (size_t)(m0 + ln) * Kd;

    for (int k0 = 0; k0 < Kd; k0 += 32) {
        // A fragment: M = ln, halves cover K run [k0+16g, +16)
        v16bf af = ldfrag(arow + k0 + 16 * lg, arow + k0 + 16 * lg + 8);
#pragma unroll
        for (int j = 0; j < 4; ++j) {
            // B fragment: N = n0+j*16+ln, halves cover K run [k0+16g, +16)
            const unsigned short* bp = Bt + (size_t)(n0 + j * 16 + ln) * Kd + k0 + 16 * lg;
            acc[j] = wmma_bf16(af, ldfrag(bp, bp + 8), acc[j]);
        }
    }

#pragma unroll
    for (int j = 0; j < 4; ++j)
#pragma unroll
        for (int r = 0; r < 8; ++r) {
            size_t idx = (size_t)(m0 + r + 8 * lg) * N + (n0 + j * 16 + ln);
            if constexpr (OUT_BF16) Cb[idx] = f2bf(acc[j][r]);
            else                    Cf[idx] = acc[j][r];
        }
}

// ---------- flash-style GQA attention ----------
// Qb: [B*T, C] bf16 ; Kb: [B*T, NKV*DH] bf16 ; Vt: [B][NKV*DH][T] bf16 ; Yb: [B*T, C] bf16
__global__ void __launch_bounds__(128, 1)
attn_wmma(const unsigned short* __restrict__ Qb,
          const unsigned short* __restrict__ Kb,
          const unsigned short* __restrict__ Vt,
          unsigned short* __restrict__ Yb) {
    __shared__ unsigned short Pl[4][16 * 32];   // per-wave P tile (16 q x 32 k) bf16

    const int lane = threadIdx.x & 31;
    const int ln = lane & 15, lg = lane >> 4;
    const int wslot = threadIdx.x >> 5;
    const int w = blockIdx.x * 4 + wslot;       // 4096 waves total
    const int qt = w & (T_SEQ / 16 - 1);        // 0..127
    const int hh = w >> 7;                      // 0..31
    const int r = hh & (RQ - 1);
    const int g = (hh >> 3) & (NKV - 1);
    const int b = hh >> 4;
    const int q0 = qt << 4;
    const int hq = g * RQ + r;
    const float scale = 0.08838834764831845f;   // 1/sqrt(128)
    const float NEG = -3.0e38f;

    // Q fragments (16 rows x 128 d, 4 chunks of k=32)
    v16bf qa[4];
    const unsigned short* qrow = Qb + (size_t)(b * T_SEQ + q0 + ln) * CDIM + hq * DH;
#pragma unroll
    for (int c = 0; c < 4; ++c)
        qa[c] = ldfrag(qrow + c * 32 + 16 * lg, qrow + c * 32 + 16 * lg + 8);

    float mrow[8], lrow[8];
    v8f o[8];
#pragma unroll
    for (int i = 0; i < 8; ++i) { mrow[i] = NEG; lrow[i] = 0.f; o[i] = v8f{}; }

    const unsigned short* kbase = Kb + (size_t)(b * T_SEQ) * (NKV * DH) + g * DH;
    const unsigned short* vbase = Vt + ((size_t)b * NKV * DH + g * DH) * T_SEQ;
    const int nkt = (q0 >> 5) + 1;              // 32-key steps covering keys <= q0+15

    for (int kt = 0; kt < nkt; ++kt) {
        const int k0 = kt << 5;
        v8f s0 = {}, s1 = {};
#pragma unroll
        for (int c = 0; c < 4; ++c) {
            // Kt B-fragment: N = key = k0(+16)+ln ; K = d = c*32 + 16*lg + h (contiguous)
            const unsigned short* kp0 = kbase + (size_t)(k0 + ln) * (NKV * DH) + c * 32 + 16 * lg;
            const unsigned short* kp1 = kbase + (size_t)(k0 + 16 + ln) * (NKV * DH) + c * 32 + 16 * lg;
            s0 = wmma_bf16(qa[c], ldfrag(kp0, kp0 + 8), s0);
            s1 = wmma_bf16(qa[c], ldfrag(kp1, kp1 + 8), s1);
        }

        // online softmax over the 32 new keys; rows live in 16-lane halves (wave32)
#pragma unroll
        for (int rr = 0; rr < 8; ++rr) {
            const int row = q0 + rr + 8 * lg;
            float x0 = s0[rr] * scale; if (k0 + ln      > row) x0 = NEG;
            float x1 = s1[rr] * scale; if (k0 + 16 + ln > row) x1 = NEG;
            float mx = fmaxf(x0, x1);
            mx = fmaxf(mx, __shfl_xor(mx, 1, 32));
            mx = fmaxf(mx, __shfl_xor(mx, 2, 32));
            mx = fmaxf(mx, __shfl_xor(mx, 4, 32));
            mx = fmaxf(mx, __shfl_xor(mx, 8, 32));
            const float mn = fmaxf(mrow[rr], mx);
            const float a  = __expf(mrow[rr] - mn);
            const float p0 = __expf(x0 - mn);
            const float p1 = __expf(x1 - mn);
            float sum = p0 + p1;
            sum += __shfl_xor(sum, 1, 32);
            sum += __shfl_xor(sum, 2, 32);
            sum += __shfl_xor(sum, 4, 32);
            sum += __shfl_xor(sum, 8, 32);
            mrow[rr] = mn;
            lrow[rr] = lrow[rr] * a + sum;
#pragma unroll
            for (int j = 0; j < 8; ++j) o[j][rr] *= a;   // rescale O rows in place
            Pl[wslot][(rr + 8 * lg) * 32 + ln]      = f2bf(p0);
            Pl[wslot][(rr + 8 * lg) * 32 + 16 + ln] = f2bf(p1);
        }

        // LDS ops of one wave complete in-order; fence issue order before re-reading.
        asm volatile("s_wait_dscnt 0" ::: "memory");

        // P A-fragment: M = ln ; K run [16g, +16)
        const unsigned short* pr = &Pl[wslot][ln * 32];
        v16bf pa = mkfrag(*(const us8*)(pr + 16 * lg), *(const us8*)(pr + 16 * lg + 8));

#pragma unroll
        for (int j = 0; j < 8; ++j) {
            // V B-fragment: N = d = j*16+ln ; K = key = k0 + 16*lg + h (contiguous in Vt)
            const unsigned short* vp = vbase + (size_t)(j * 16 + ln) * T_SEQ + k0 + 16 * lg;
            o[j] = wmma_bf16(pa, ldfrag(vp, vp + 8), o[j]);
        }
    }

    unsigned short* yrow = Yb + (size_t)(b * T_SEQ) * CDIM + hq * DH;
#pragma unroll
    for (int j = 0; j < 8; ++j)
#pragma unroll
        for (int rr = 0; rr < 8; ++rr) {
            int row = q0 + rr + 8 * lg;
            yrow[(size_t)row * CDIM + j * 16 + ln] = f2bf(o[j][rr] / lrow[rr]);
        }
}

// ---------- host launcher ----------
extern "C" void kernel_launch(void* const* d_in, const int* in_sizes, int n_in,
                              void* d_out, int out_size, void* d_ws, size_t ws_size,
                              hipStream_t stream) {
    const float* x  = (const float*)d_in[0];
    const float* Wq = (const float*)d_in[1];
    const float* Wk = (const float*)d_in[2];
    const float* Wv = (const float*)d_in[3];
    const float* Wo = (const float*)d_in[4];

    char* ws = (char*)d_ws;
    unsigned short* xb  = (unsigned short*)(ws + 0);          // 16,777,216 B
    unsigned short* WqT = (unsigned short*)(ws + 16777216);   //  8,388,608 B
    unsigned short* WkT = (unsigned short*)(ws + 25165824);   //  1,048,576 B
    unsigned short* WvT = (unsigned short*)(ws + 26214400);   //  1,048,576 B
    unsigned short* WoT = (unsigned short*)(ws + 27262976);   //  8,388,608 B
    unsigned short* Qb  = (unsigned short*)(ws + 35651584);   // 16,777,216 B
    unsigned short* Kb  = (unsigned short*)(ws + 52428800);   //  2,097,152 B
    unsigned short* Vb  = (unsigned short*)(ws + 54525952);   //  2,097,152 B
    unsigned short* Vt  = (unsigned short*)(ws + 56623104);   //  2,097,152 B
    unsigned short* Yb  = (unsigned short*)(ws + 58720256);   // 16,777,216 B  (total 72 MiB)

    const int M = BB * T_SEQ;                                 // 4096

    cvt_bf16<<<(M * CDIM + 255) / 256, 256, 0, stream>>>(x, xb, M * CDIM);

    dim3 tq(CDIM / 32, CDIM / 32);                            // [2048,2048] transpose
    dim3 tk(CDIM / 32, (NKV * DH) / 32);                      // [2048,256] transpose
    transpose_cvt<<<tq, 256, 0, stream>>>(Wq, WqT, CDIM, CDIM);
    transpose_cvt<<<tk, 256, 0, stream>>>(Wk, WkT, CDIM, NKV * DH);
    transpose_cvt<<<tk, 256, 0, stream>>>(Wv, WvT, CDIM, NKV * DH);
    transpose_cvt<<<tq, 256, 0, stream>>>(Wo, WoT, CDIM, CDIM);

    // Q = x @ Wq : 4096x2048x2048 ; waves = 256*32 = 8192 -> 1024 blocks of 8 waves
    gemm_wmma<true><<<1024, 256, 0, stream>>>(xb, WqT, nullptr, Qb, M, CDIM, CDIM);
    // K,V = x @ Wk/Wv : 4096x256x2048 ; waves = 256*4 = 1024 -> 128 blocks
    gemm_wmma<true><<<128, 256, 0, stream>>>(xb, WkT, nullptr, Kb, M, NKV * DH, CDIM);
    gemm_wmma<true><<<128, 256, 0, stream>>>(xb, WvT, nullptr, Vb, M, NKV * DH, CDIM);

    dim3 tv(T_SEQ / 32, (NKV * DH) / 32, BB);
    vtrans<<<tv, 256, 0, stream>>>(Vb, Vt);

    // flash attention: 4096 waves, 4 per block
    attn_wmma<<<1024, 128, 0, stream>>>(Qb, Kb, Vt, Yb);

    // out = Y @ Wo -> fp32 d_out
    gemm_wmma<false><<<1024, 256, 0, stream>>>(Yb, WoT, (float*)d_out, nullptr, M, CDIM, CDIM);
}
